// MoE_23656679867558
// MI455X (gfx1250) — matrix operand order repaired and checked by reading, not compiled, not run on hardware
//
#include <hip/hip_runtime.h>
#include <math.h>

#define NEXP  8
#define CAP   4096
#define NTOK  16384
#define DIM   1024
#define HID   2048
#define NBINS 16384

typedef _Float16 h16;
typedef __attribute__((ext_vector_type(16))) _Float16 v16h;
typedef __attribute__((ext_vector_type(8)))  _Float16 v8h;
typedef __attribute__((ext_vector_type(4)))  _Float16 v4h;
typedef __attribute__((ext_vector_type(8)))  float    v8f;

#define WMMA16(a,b,c) __builtin_amdgcn_wmma_f32_16x16x32_f16(false,(a),false,(b),(short)0,(c),false,false)

// gfx1250 async global->LDS copy (GV mode): VDST = per-lane LDS byte offset,
// VADDR = per-lane 64-bit global address. Tracked by ASYNCcnt.
#define ASYNC_B128(ldsoff, gaddr) \
  asm volatile("global_load_async_to_lds_b128 %0, %1, off" :: "v"(ldsoff), "v"(gaddr) : "memory")
#define WAIT_ASYNC() asm volatile("s_wait_asynccnt 0x0" ::: "memory")

// Low 32 bits of a generic LDS pointer = workgroup-relative LDS byte offset
// (LDS aperture match uses addr[63:32] only; HW truncates to addr[31:0]).
static __device__ __forceinline__ unsigned lds_u32(const void* p) {
  return (unsigned)(unsigned long long)p;
}

// ---------------- WMMA fragment loaders (per ISA 16-bit layouts) ----------------
// A tile in LDS: row-major [m][k], stride 32. Lanes 0-15: M=lane, K=0-7 & 16-23;
// lanes 16-31: M=lane-16, K=8-15 & 24-31.
__device__ __forceinline__ v16h frag_a(const h16* s, int mo, int lane) {
  const int hl = lane >> 4, row = lane & 15;
  const h16* p = s + (mo + row) * 32 + hl * 8;
  v8h lo = *(const v8h*)p;          // K = hl*8 + 0..7
  v8h hi = *(const v8h*)(p + 16);   // K = 16 + hl*8 + 0..7
  v16h r;
  #pragma unroll
  for (int i = 0; i < 8; ++i) { r[i] = lo[i]; r[i + 8] = hi[i]; }
  return r;
}
// B tile in LDS: [n][k], stride 32. Lane = column n, elements = K=(lane/16)*8+0..7 & 16+(lane/16)*8+0..7.
__device__ __forceinline__ v16h frag_b(const h16* s, int no, int lane) {
  const int hl = lane >> 4, col = lane & 15;
  const h16* p = s + (no + col) * 32 + hl * 8;
  v8h lo = *(const v8h*)p;
  v8h hi = *(const v8h*)(p + 16);
  v16h r;
  #pragma unroll
  for (int i = 0; i < 8; ++i) { r[i] = lo[i]; r[i + 8] = hi[i]; }
  return r;
}

// ---------------- utility kernels ----------------
__global__ void moe_zero_f32(float* __restrict__ p, long long n) {
  long long i = (long long)blockIdx.x * blockDim.x + threadIdx.x;
  const long long stride = (long long)gridDim.x * blockDim.x;
  for (; i < n; i += stride) p[i] = 0.f;
}

// Transpose-convert: src (E, K, N) f32  ->  dst (E, N, K) f16. LDS-tiled, coalesced.
__global__ __launch_bounds__(256) void moe_cvt_t(const float* __restrict__ src,
                                                 h16* __restrict__ dst, int K, int N) {
  __shared__ float tile[32][33];
  const int e  = blockIdx.z;
  const int n0 = blockIdx.x * 32, k0 = blockIdx.y * 32;
  const int tx = threadIdx.x & 31, ty = threadIdx.x >> 5;  // ty: 0..7
  const float* s = src + (size_t)e * K * N;
  h16* d = dst + (size_t)e * K * N;
  #pragma unroll
  for (int r = 0; r < 4; ++r) {
    const int k = ty * 4 + r;
    tile[k][tx] = s[(size_t)(k0 + k) * N + n0 + tx];
  }
  __syncthreads();
  #pragma unroll
  for (int r = 0; r < 4; ++r) {
    const int n = ty * 4 + r;
    d[(size_t)(n0 + n) * K + k0 + tx] = (h16)tile[tx][n];
  }
}

// ---------------- router: scores_t[e][n] = softmax(x @ wg)[n][e] ----------------
__global__ __launch_bounds__(256) void moe_router(const float* __restrict__ x,
                                                  const float* __restrict__ wg,
                                                  float* __restrict__ scores_t) {
  const int lane = threadIdx.x & 31;
  const int wave = threadIdx.x >> 5;
  const int tok  = blockIdx.x * 8 + wave;
  const float* xp = x + (size_t)tok * DIM;
  float acc[NEXP];
  #pragma unroll
  for (int e = 0; e < NEXP; ++e) acc[e] = 0.f;
  for (int d = lane; d < DIM; d += 32) {
    const float xv = xp[d];
    const float4 g0 = *(const float4*)(wg + d * NEXP);
    const float4 g1 = *(const float4*)(wg + d * NEXP + 4);
    acc[0] += xv * g0.x; acc[1] += xv * g0.y; acc[2] += xv * g0.z; acc[3] += xv * g0.w;
    acc[4] += xv * g1.x; acc[5] += xv * g1.y; acc[6] += xv * g1.z; acc[7] += xv * g1.w;
  }
  #pragma unroll
  for (int e = 0; e < NEXP; ++e) {
    float v = acc[e];
    #pragma unroll
    for (int off = 16; off > 0; off >>= 1) v += __shfl_xor(v, off, 32);
    acc[e] = v;
  }
  if (lane == 0) {
    float m = acc[0];
    #pragma unroll
    for (int e = 1; e < NEXP; ++e) m = fmaxf(m, acc[e]);
    float ex[NEXP]; float sum = 0.f;
    #pragma unroll
    for (int e = 0; e < NEXP; ++e) { ex[e] = __expf(acc[e] - m); sum += ex[e]; }
    const float inv = 1.f / sum;
    #pragma unroll
    for (int e = 0; e < NEXP; ++e) scores_t[(size_t)e * NTOK + tok] = ex[e] * inv;
  }
}

// ---------------- expert-choice top-CAP via histogram threshold ----------------
__device__ __forceinline__ int score_bin(float s) {
  int b = (int)(s * (float)NBINS);
  return b < 0 ? 0 : (b > NBINS - 1 ? NBINS - 1 : b);
}

__global__ void moe_hist(const float* __restrict__ scores_t, unsigned* __restrict__ hist) {
  const int n = blockIdx.x * blockDim.x + threadIdx.x;
  #pragma unroll
  for (int e = 0; e < NEXP; ++e) {
    const float s = scores_t[(size_t)e * NTOK + n];
    atomicAdd(&hist[e * NBINS + score_bin(s)], 1u);
  }
}

__global__ __launch_bounds__(256) void moe_select(const unsigned* __restrict__ hist,
                                                  int* __restrict__ thresh,
                                                  int* __restrict__ cntgt) {
  __shared__ unsigned csum[256];
  const int e = blockIdx.x, t = threadIdx.x;
  const unsigned* h = hist + (size_t)e * NBINS;
  const int CH = NBINS / 256;
  unsigned s = 0;
  for (int i = 0; i < CH; ++i) s += h[t * CH + i];
  csum[t] = s;
  __syncthreads();
  if (t == 0) {
    unsigned run = 0; int tb = 0; unsigned cg = 0;
    for (int c = 255; c >= 0; --c) {
      const unsigned nr = run + csum[c];
      if (nr >= CAP) {
        unsigned r2 = run;
        for (int b = c * CH + CH - 1; b >= c * CH; --b) {
          const unsigned nb = r2 + h[b];
          if (nb >= CAP) { tb = b; cg = r2; break; }
          r2 = nb;
        }
        break;
      }
      run = nr;
    }
    thresh[e] = tb;
    cntgt[e]  = (int)cg;
  }
}

__global__ void moe_compact(const float* __restrict__ scores_t, const int* __restrict__ thresh,
                            int* __restrict__ cntA, int* __restrict__ idxb, float* __restrict__ scrb) {
  const int n = blockIdx.x * blockDim.x + threadIdx.x;
  const int e = blockIdx.y;
  const float s = scores_t[(size_t)e * NTOK + n];
  if (score_bin(s) > thresh[e]) {
    const int p = atomicAdd(&cntA[e], 1);
    idxb[e * CAP + p] = n;
    scrb[e * CAP + p] = s;
  }
}

__global__ void moe_fill_ties(const float* __restrict__ scores_t, const int* __restrict__ thresh,
                              const int* __restrict__ cntgt, int* __restrict__ idxb,
                              float* __restrict__ scrb) {
  const int e = blockIdx.x;
  if (threadIdx.x != 0) return;
  const int t = thresh[e];
  int c = cntgt[e];
  for (int n = 0; n < NTOK && c < CAP; ++n) {
    const float s = scores_t[(size_t)e * NTOK + n];
    if (score_bin(s) == t) { idxb[e * CAP + c] = n; scrb[e * CAP + c] = s; ++c; }
  }
}

// ---------------- dispatch: routed[e][c][:] = f16(x[idx] * score) ----------------
__global__ __launch_bounds__(256) void moe_dispatch(const float* __restrict__ x,
                                                    const int* __restrict__ idxb,
                                                    const float* __restrict__ scrb,
                                                    h16* __restrict__ routed) {
  const int c = blockIdx.x, e = blockIdx.y;
  const int row = idxb[e * CAP + c];
  const float sc = scrb[e * CAP + c];
  const float* xp = x + (size_t)row * DIM;
  h16* rp = routed + ((size_t)(e * CAP + c)) * DIM;
  const int d = threadIdx.x * 4;
  const float4 v = *(const float4*)(xp + d);
  v4h o = { (h16)(v.x * sc), (h16)(v.y * sc), (h16)(v.z * sc), (h16)(v.w * sc) };
  *(v4h*)(rp + d) = o;
}

// ---------------- GEMM1: h = silu(A@W1) * (A@W3); async-LDS double buffered ----------------
__global__ __launch_bounds__(256) void moe_gemm1(const h16* __restrict__ A,
                                                 const h16* __restrict__ W1t,
                                                 const h16* __restrict__ W3t,
                                                 h16* __restrict__ H) {
  __shared__ __align__(16) h16 sA [2 * 128 * 32];
  __shared__ __align__(16) h16 sB1[2 * 64 * 32];
  __shared__ __align__(16) h16 sB3[2 * 64 * 32];
  const int e  = blockIdx.z;
  const int bm = blockIdx.y;   // 128-row tile
  const int bn = blockIdx.x;   // 64-col tile
  const int tid = threadIdx.x, lane = tid & 31, wave = tid >> 5;
  const int mo = (wave >> 1) * 32, no = (wave & 1) * 32;

  v8f acc1[2][2] = {};
  v8f acc3[2][2] = {};

  const h16* Ab  = A   + ((size_t)(e * CAP + bm * 128)) * DIM;
  const h16* B1b = W1t + ((size_t)(e * HID + bn * 64)) * DIM;   // N-major: [n][k]
  const h16* B3b = W3t + ((size_t)(e * HID + bn * 64)) * DIM;

  const int arow = tid >> 1, ak  = (tid & 1) * 16;  // A: 128 x 32
  const int brow = tid >> 2, bk0 = (tid & 3) * 8;   // B: 64 x 32 (n-major)

  const unsigned aA  = lds_u32(sA);
  const unsigned aB1 = lds_u32(sB1);
  const unsigned aB3 = lds_u32(sB3);

  // prologue: prefetch K-tile 0 into buffer 0
  {
    const unsigned long long ga = (unsigned long long)(Ab + (size_t)arow * DIM + ak);
    ASYNC_B128(aA + tid * 32,      ga);
    ASYNC_B128(aA + tid * 32 + 16, ga + 16);
    ASYNC_B128(aB1 + tid * 16, (unsigned long long)(B1b + (size_t)brow * DIM + bk0));
    ASYNC_B128(aB3 + tid * 16, (unsigned long long)(B3b + (size_t)brow * DIM + bk0));
  }

  int db = 0;
  for (int kk = 0; kk < DIM; kk += 32) {
    WAIT_ASYNC();
    __syncthreads();                         // current buffer ready for all waves
    if (kk + 32 < DIM) {                     // prefetch next tile into other buffer
      const int nb = db ^ 1, nk = kk + 32;
      const unsigned long long ga = (unsigned long long)(Ab + (size_t)arow * DIM + nk + ak);
      ASYNC_B128(aA + nb * 8192 + tid * 32,      ga);
      ASYNC_B128(aA + nb * 8192 + tid * 32 + 16, ga + 16);
      ASYNC_B128(aB1 + nb * 4096 + tid * 16, (unsigned long long)(B1b + (size_t)brow * DIM + nk + bk0));
      ASYNC_B128(aB3 + nb * 4096 + tid * 16, (unsigned long long)(B3b + (size_t)brow * DIM + nk + bk0));
    }
    const h16* cA  = sA  + db * 4096;
    const h16* cB1 = sB1 + db * 2048;
    const h16* cB3 = sB3 + db * 2048;
    const v16h a0 = frag_a(cA, mo,      lane);
    const v16h a1 = frag_a(cA, mo + 16, lane);
    const v16h p0 = frag_b(cB1, no,      lane);
    const v16h p1 = frag_b(cB1, no + 16, lane);
    const v16h q0 = frag_b(cB3, no,      lane);
    const v16h q1 = frag_b(cB3, no + 16, lane);
    acc1[0][0] = WMMA16(a0, p0, acc1[0][0]);
    acc1[0][1] = WMMA16(a0, p1, acc1[0][1]);
    acc1[1][0] = WMMA16(a1, p0, acc1[1][0]);
    acc1[1][1] = WMMA16(a1, p1, acc1[1][1]);
    acc3[0][0] = WMMA16(a0, q0, acc3[0][0]);
    acc3[0][1] = WMMA16(a0, q1, acc3[0][1]);
    acc3[1][0] = WMMA16(a1, q0, acc3[1][0]);
    acc3[1][1] = WMMA16(a1, q1, acc3[1][1]);
    db ^= 1;
  }

  const int hl = lane >> 4, col = lane & 15;
  h16* Hb = H + ((size_t)(e * CAP + bm * 128)) * HID + (size_t)bn * 64;
  #pragma unroll
  for (int fi = 0; fi < 2; ++fi)
    #pragma unroll
    for (int fj = 0; fj < 2; ++fj)
      #pragma unroll
      for (int i = 0; i < 8; ++i) {
        const float g = acc1[fi][fj][i];
        const float u = acc3[fi][fj][i];
        const float a = (g / (1.f + __expf(-g))) * u;   // silu(g) * u
        const int m = mo + fi * 16 + hl * 8 + i;
        const int n = no + fj * 16 + col;
        Hb[(size_t)m * HID + n] = (h16)a;
      }
}

// ---------------- GEMM2: out += scatter(H @ W2); async-LDS double buffered ----------------
__global__ __launch_bounds__(256) void moe_gemm2(const h16* __restrict__ H,
                                                 const h16* __restrict__ W2t,
                                                 const int* __restrict__ idxb,
                                                 float* __restrict__ out) {
  __shared__ __align__(16) h16 sA[2 * 128 * 32];
  __shared__ __align__(16) h16 sB[2 * 64 * 32];
  const int e  = blockIdx.z;
  const int bm = blockIdx.y;   // 128 capacity slots
  const int bn = blockIdx.x;   // 64 dims
  const int tid = threadIdx.x, lane = tid & 31, wave = tid >> 5;
  const int mo = (wave >> 1) * 32, no = (wave & 1) * 32;

  v8f acc[2][2] = {};

  const h16* Ab = H   + ((size_t)(e * CAP + bm * 128)) * HID;
  const h16* Bb = W2t + ((size_t)(e * DIM + bn * 64)) * HID;   // N-major: [n][k]

  const int arow = tid >> 1, ak  = (tid & 1) * 16;
  const int brow = tid >> 2, bk0 = (tid & 3) * 8;

  const unsigned aA = lds_u32(sA);
  const unsigned aB = lds_u32(sB);

  {
    const unsigned long long ga = (unsigned long long)(Ab + (size_t)arow * HID + ak);
    ASYNC_B128(aA + tid * 32,      ga);
    ASYNC_B128(aA + tid * 32 + 16, ga + 16);
    ASYNC_B128(aB + tid * 16, (unsigned long long)(Bb + (size_t)brow * HID + bk0));
  }

  int db = 0;
  for (int kk = 0; kk < HID; kk += 32) {
    WAIT_ASYNC();
    __syncthreads();
    if (kk + 32 < HID) {
      const int nb = db ^ 1, nk = kk + 32;
      const unsigned long long ga = (unsigned long long)(Ab + (size_t)arow * HID + nk + ak);
      ASYNC_B128(aA + nb * 8192 + tid * 32,      ga);
      ASYNC_B128(aA + nb * 8192 + tid * 32 + 16, ga + 16);
      ASYNC_B128(aB + nb * 4096 + tid * 16, (unsigned long long)(Bb + (size_t)brow * HID + nk + bk0));
    }
    const h16* cA = sA + db * 4096;
    const h16* cB = sB + db * 2048;
    const v16h a0 = frag_a(cA, mo,      lane);
    const v16h a1 = frag_a(cA, mo + 16, lane);
    const v16h b0 = frag_b(cB, no,      lane);
    const v16h b1 = frag_b(cB, no + 16, lane);
    acc[0][0] = WMMA16(a0, b0, acc[0][0]);
    acc[0][1] = WMMA16(a0, b1, acc[0][1]);
    acc[1][0] = WMMA16(a1, b0, acc[1][0]);
    acc[1][1] = WMMA16(a1, b1, acc[1][1]);
    db ^= 1;
  }

  const int hl = lane >> 4, col = lane & 15;
  const int* ib = idxb + e * CAP;
  #pragma unroll
  for (int fi = 0; fi < 2; ++fi)
    #pragma unroll
    for (int fj = 0; fj < 2; ++fj)
      #pragma unroll
      for (int i = 0; i < 8; ++i) {
        const int m = bm * 128 + mo + fi * 16 + hl * 8 + i;
        const int tok = ib[m];
        const int n = bn * 64 + no + fj * 16 + col;
        atomicAdd(&out[(size_t)tok * DIM + n], acc[fi][fj][i]);
      }
}

// ---------------- host launcher ----------------
extern "C" void kernel_launch(void* const* d_in, const int* in_sizes, int n_in,
                              void* d_out, int out_size, void* d_ws, size_t ws_size,
                              hipStream_t stream) {
  const float* x  = (const float*)d_in[0];
  const float* wg = (const float*)d_in[1];
  const float* w1 = (const float*)d_in[2];
  const float* w3 = (const float*)d_in[3];
  const float* w2 = (const float*)d_in[4];
  float* out = (float*)d_out;
  (void)in_sizes; (void)n_in; (void)ws_size;

  char* base = (char*)d_ws;
  size_t off = 0;
  auto carve = [&](size_t bytes) -> char* {
    char* r = base + off;
    off = (off + bytes + 255) & ~(size_t)255;
    return r;
  };
  float*    scores = (float*)   carve((size_t)NEXP * NTOK * 4);
  unsigned* hist   = (unsigned*)carve((size_t)NEXP * NBINS * 4);
  int*      thresh = (int*)     carve(NEXP * 4);
  int*      cntgt  = (int*)     carve(NEXP * 4);
  int*      cntA   = (int*)     carve(NEXP * 4);
  int*      idxb   = (int*)     carve((size_t)NEXP * CAP * 4);
  float*    scrb   = (float*)   carve((size_t)NEXP * CAP * 4);
  h16*      routed = (h16*)     carve((size_t)NEXP * CAP * DIM * 2);
  h16*      hbuf   = (h16*)     carve((size_t)NEXP * CAP * HID * 2);
  h16*      w1t    = (h16*)     carve((size_t)NEXP * DIM * HID * 2);
  h16*      w3t    = (h16*)     carve((size_t)NEXP * DIM * HID * 2);
  h16*      w2t    = (h16*)     carve((size_t)NEXP * HID * DIM * 2);

  // init: zero output, histogram, counters (thresh/cntgt/cntA = 3 contiguous 256B chunks)
  moe_zero_f32<<<2048, 256, 0, stream>>>(out, (long long)out_size);
  moe_zero_f32<<<256, 256, 0, stream>>>((float*)hist, (long long)NEXP * NBINS);
  moe_zero_f32<<<1, 256, 0, stream>>>((float*)thresh, 192);

  // router + expert-choice selection
  moe_router<<<NTOK / 8, 256, 0, stream>>>(x, wg, scores);
  moe_hist<<<NTOK / 256, 256, 0, stream>>>(scores, hist);
  moe_select<<<NEXP, 256, 0, stream>>>(hist, thresh, cntgt);
  moe_compact<<<dim3(NTOK / 256, NEXP), 256, 0, stream>>>(scores, thresh, cntA, idxb, scrb);
  moe_fill_ties<<<NEXP, 32, 0, stream>>>(scores, thresh, cntgt, idxb, scrb);

  // dispatch + transposed f16 weights (w1t/w3t: (E,HID,DIM), w2t: (E,DIM,HID))
  moe_dispatch<<<dim3(CAP, NEXP), 256, 0, stream>>>(x, idxb, scrb, routed);
  moe_cvt_t<<<dim3(HID / 32, DIM / 32, NEXP), 256, 0, stream>>>(w1, w1t, DIM, HID);
  moe_cvt_t<<<dim3(HID / 32, DIM / 32, NEXP), 256, 0, stream>>>(w3, w3t, DIM, HID);
  moe_cvt_t<<<dim3(DIM / 32, HID / 32, NEXP), 256, 0, stream>>>(w2, w2t, HID, DIM);

  // expert FFN (WMMA) + combine
  moe_gemm1<<<dim3(HID / 64, CAP / 128, NEXP), 256, 0, stream>>>(routed, w1t, w3t, hbuf);
  moe_gemm2<<<dim3(DIM / 64, CAP / 128, NEXP), 256, 0, stream>>>(hbuf, w2t, idxb, out);
}
